// ACE_NODE_24653112279535
// MI455X (gfx1250) — matrix operand order repaired and checked by reading, not compiled, not run on hardware
//
#include <hip/hip_runtime.h>
#include <stdint.h>
#include <stddef.h>

// ---------------- problem constants ----------------
#define HDIM   64
#define INPD   40
#define BATCH  128
#define SEQ    128
#define NSTEP  8
#define MT     16              // batch rows per group == WMMA M tile
#define NGRP   (BATCH / MT)    // 8 independent ODE chains
#define NTHR   512             // 16 wave32 waves per workgroup
#define HS     (1.0f / 8.0f)   // fixed Tsit5 substep

#define LOG2E  1.4426950408889634f
#define LN2    0.6931471805599453f

typedef __attribute__((ext_vector_type(16))) __bf16         v16bf;
typedef __attribute__((ext_vector_type(16))) unsigned short v16us;
typedef __attribute__((ext_vector_type(8)))  float          v8f;

// ---------------- Tsit5 tableau ----------------
__constant__ float STG_C[6] = {0.0f, 0.161f, 0.327f, 0.9f, 0.9800255409045097f, 1.0f};
__constant__ float STG_A[6][5] = {
  {0.f,0.f,0.f,0.f,0.f},
  {0.161f,0.f,0.f,0.f,0.f},
  {-0.008480655492356989f, 0.335480655492357f, 0.f,0.f,0.f},
  {2.8971530571054935f, -6.359448489975075f, 4.3622954328695815f, 0.f,0.f},
  {5.325864828439257f, -11.748883564062828f, 7.4955393428898365f, -0.09249506636175525f, 0.f},
  {5.86145544294642f, -12.92096931784711f, 8.159367898576159f, -0.071584973281401f, -0.028269050394068383f}
};
__constant__ float STG_B[6] = {0.09646076681806523f, 0.01f, 0.4798896504144996f,
                               1.379008574103742f, -3.290069515436081f, 2.324710524099774f};

// ---------------- fast transcendentals (raw v_exp_f32 / v_log_f32 / v_rcp_f32) ----------
static __device__ __forceinline__ float fexp2(float x) {
#if __has_builtin(__builtin_amdgcn_exp2f)
  return __builtin_amdgcn_exp2f(x);
#else
  return __expf(x * LN2);
#endif
}
static __device__ __forceinline__ float flog2(float x) {
#if __has_builtin(__builtin_amdgcn_logf)
  return __builtin_amdgcn_logf(x);
#else
  return __logf(x) * LOG2E;
#endif
}
static __device__ __forceinline__ float frcp(float x) {
#if __has_builtin(__builtin_amdgcn_rcpf)
  return __builtin_amdgcn_rcpf(x);
#else
  return 1.f / x;
#endif
}
// branchless, numerically stable: softplus(x) = max(x,0) + ln2*log2(1 + 2^(-|x|*log2e))
static __device__ __forceinline__ float softplus_f(float x) {
  float e = fexp2(-fabsf(x) * LOG2E);
  return fmaxf(x, 0.f) + LN2 * flog2(1.f + e);
}
static __device__ __forceinline__ float sigmoid_f(float x) {
  return frcp(1.f + fexp2(-x * LOG2E));
}
static __device__ __forceinline__ float tanh_f(float x) {
  x = fminf(15.f, fmaxf(-15.f, x));
  float e = fexp2(2.f * x * LOG2E);
  return (e - 1.f) * frcp(e + 1.f);
}

static __device__ __forceinline__ unsigned short f2bf(float f) {
  unsigned u = __float_as_uint(f);
  u = (u + 0x7FFFu + ((u >> 16) & 1u)) >> 16;   // round-to-nearest-even
  return (unsigned short)u;
}

static __device__ __forceinline__ v8f wmma_bf16(v16bf a, v16bf b, v8f c) {
  // D(16x16,f32) = A(16x32,bf16) * B(32x16,bf16) + C
  return __builtin_amdgcn_wmma_f32_16x16x32_bf16(false, a, false, b, (short)0, c, false, false);
}

// A-matrix (16-bit, 16x32) lane/element mapping per CDNA5 ISA 7.12.2:
//  lanes 0-15: M=lane, elems 0..7 -> K = kbase+0..7, elems 8..15 -> K = 16+kbase+0..7
//  lanes 16-31: same M, kbase = 8
// Both 8-element halves are contiguous in K -> compiler emits 2x ds_load_b128 per tile.
static __device__ __forceinline__ v16bf load_A_tile(const unsigned short* __restrict__ buf,
                                                    int ld, int kt) {
  int l  = threadIdx.x & 31;
  int m  = l & 15;
  int kb = (l >> 4) * 8;
  v16us a;
  #pragma unroll
  for (int e = 0; e < 16; ++e) {
    int k = kt * 32 + ((e < 8) ? (kb + e) : (16 + kb + (e - 8)));
    a[e] = buf[m * ld + k];
  }
  return __builtin_bit_cast(v16bf, a);
}

// B tiles pre-packed lane-contiguous: tile=(nt*ktiles+kt), entry = tile*512 + lane*16 + e
static __device__ __forceinline__ v16bf load_B_tile(const unsigned short* __restrict__ wp,
                                                    int ktiles, int nt, int kt) {
  const v16bf* p = (const v16bf*)wp;
  return p[(nt * ktiles + kt) * 32 + (threadIdx.x & 31)];
}

// ---------------- weight packing (f32 row-major (N,K) -> bf16 WMMA B-tiles) ----------------
__global__ void pack_weight(const float* __restrict__ W, unsigned short* __restrict__ dst,
                            int Kreal, int ktiles, int ntiles) {
  int total = ktiles * ntiles * 512;
  for (int idx = blockIdx.x * blockDim.x + threadIdx.x; idx < total;
       idx += gridDim.x * blockDim.x) {
    int tile = idx >> 9;
    int w    = idx & 511;
    int l = w >> 4, e = w & 15;
    int kt = tile % ktiles, nt = tile / ktiles;
    int k = kt * 32 + ((e < 8) ? ((l >> 4) * 8 + e) : (16 + (l >> 4) * 8 + (e - 8)));   // B-matrix (32x16, 16-bit) lane layout
    int n = nt * 16 + (l & 15);
    float v = (k < Kreal) ? W[n * Kreal + k] : 0.f;
    dst[idx] = f2bf(v);
  }
}

// ---------------- params ----------------
struct AceParams {
  const float* __restrict__ x_seq;
  const float* __restrict__ fb0; const float* __restrict__ fb1;
  const float* __restrict__ fb2; const float* __restrict__ fb3;
  const float* __restrict__ gb0; const float* __restrict__ gb1;
  const float* __restrict__ gb2; const float* __restrict__ gb3;
  const float* __restrict__ gru_wi; const float* __restrict__ gru_wh;
  const float* __restrict__ gru_b;  const float* __restrict__ gru_bn;
  const unsigned short* __restrict__ wF0; const unsigned short* __restrict__ wF1;
  const unsigned short* __restrict__ wF2; const unsigned short* __restrict__ wF3;
  const unsigned short* __restrict__ wG0; const unsigned short* __restrict__ wG1;
  const unsigned short* __restrict__ wG2; const unsigned short* __restrict__ wG3;
  float* __restrict__ yS;   // [NGRP][MT][H]
  float* __restrict__ aS;   // [NGRP][MT][H*H]
  float* __restrict__ asB;  // stage buffer, same shape as aS
  float* __restrict__ kyB;  // [NGRP][6][MT*H]
  float* __restrict__ kaB;  // [NGRP][6][MT*H*H]
  float* __restrict__ out;  // [B][S][H]
};

// two dense layers (f and g) in parallel on waves 0..7, softplus fused.
// wave id passed in as an SGPR-uniform value so all pointer selects stay scalar.
static __device__ __forceinline__ void dense_pair_layer(
    int wv,
    const unsigned short* srcF, const unsigned short* srcG, int ld, int ktiles,
    const unsigned short* wF, const unsigned short* wG,
    const float* bF, const float* bG,
    unsigned short* dstF, unsigned short* dstG) {
  if (wv < 8) {
    bool isf = wv < 4;
    int nt = wv & 3;
    const unsigned short* src = isf ? srcF : srcG;
    const unsigned short* wp  = isf ? wF : wG;
    const float* bias         = isf ? bF : bG;
    unsigned short* dst       = isf ? dstF : dstG;
    v8f c = {0.f,0.f,0.f,0.f,0.f,0.f,0.f,0.f};
    for (int kt = 0; kt < ktiles; ++kt) {
      v16bf a = load_A_tile(src, ld, kt);
      v16bf b = load_B_tile(wp, ktiles, nt, kt);
      c = wmma_bf16(a, b, c);
    }
    int l = threadIdx.x & 31;
    int n = nt * 16 + (l & 15);
    float bi = bias[n];
    #pragma unroll
    for (int r = 0; r < 8; ++r) {
      int mrow = (l >> 4) * 8 + r;
      dst[mrow * HDIM + n] = f2bf(softplus_f(c[r] + bi));
    }
  }
}

// output layers: gW3 (16x64 -> 16x4096) across all 16 waves; fW3 on waves 0..3
static __device__ __forceinline__ void out_layers(
    int wv,
    const unsigned short* __restrict__ srcF, const unsigned short* __restrict__ srcG,
    const unsigned short* __restrict__ wF3, const unsigned short* __restrict__ wG3,
    const float* __restrict__ fb3, const float* __restrict__ gb3,
    float* __restrict__ kyOut, float* __restrict__ kaOut) {
  int l = threadIdx.x & 31;
  v16bf a0 = load_A_tile(srcG, HDIM, 0);
  v16bf a1 = load_A_tile(srcG, HDIM, 1);
  for (int tt = 0; tt < 16; ++tt) {
    int nt = wv * 16 + tt;                 // uniform per wave
    if (tt + 1 < 16) {                     // prefetch next B tile pair (global_prefetch_b8)
      __builtin_prefetch((const void*)((const v16bf*)wG3 + ((nt + 1) * 2) * 32), 0, 1);
    }
    v16bf b0 = load_B_tile(wG3, 2, nt, 0);
    v16bf b1 = load_B_tile(wG3, 2, nt, 1);
    v8f c = {0.f,0.f,0.f,0.f,0.f,0.f,0.f,0.f};
    c = wmma_bf16(a0, b0, c);
    c = wmma_bf16(a1, b1, c);
    int n = nt * 16 + (l & 15);
    float bi = gb3[n];
    #pragma unroll
    for (int r = 0; r < 8; ++r) {
      int mrow = (l >> 4) * 8 + r;
      kaOut[mrow * (HDIM * HDIM) + n] = c[r] + bi;
    }
  }
  if (wv < 4) {
    int nt = wv;
    v16bf af0 = load_A_tile(srcF, HDIM, 0);
    v16bf af1 = load_A_tile(srcF, HDIM, 1);
    v16bf b0  = load_B_tile(wF3, 2, nt, 0);
    v16bf b1  = load_B_tile(wF3, 2, nt, 1);
    v8f c = {0.f,0.f,0.f,0.f,0.f,0.f,0.f,0.f};
    c = wmma_bf16(af0, b0, c);
    c = wmma_bf16(af1, b1, c);
    int n = nt * 16 + (l & 15);
    float bi = fb3[n];
    #pragma unroll
    for (int r = 0; r < 8; ++r) {
      int mrow = (l >> 4) * 8 + r;
      kyOut[mrow * HDIM + n] = c[r] + bi;
    }
  }
}

// ---------------- main persistent kernel: one workgroup == 16 batch elements, full sequence ----
__global__ __launch_bounds__(NTHR, 1)
void ace_node_main(AceParams p) {
  __shared__ unsigned short inpL[MT * 96];        // bf16 inp (yp | t | zero-pad to K=96)
  __shared__ unsigned short hfL[2][MT * HDIM];    // bf16 f-hidden ping-pong
  __shared__ unsigned short hgL[2][MT * HDIM];    // bf16 g-hidden ping-pong
  __shared__ float ysL[MT * HDIM];                // f32 stage y

  const int g   = blockIdx.x;
  const int tid = threadIdx.x;
  const int wv  = __builtin_amdgcn_readfirstlane(tid >> 5);  // SGPR-uniform wave id
  const int NA  = MT * HDIM * HDIM;  // 65536
  const int NY  = MT * HDIM;         // 1024

  float* __restrict__ yS  = p.yS  + (size_t)g * NY;
  float* __restrict__ aS  = p.aS  + (size_t)g * NA;
  float* __restrict__ asB = p.asB + (size_t)g * NA;
  float* __restrict__ kyB = p.kyB + (size_t)g * 6 * NY;
  float* __restrict__ kaB = p.kaB + (size_t)g * 6 * NA;

  for (int s = 0; s < SEQ; ++s) {
    float tb = (float)s;
    for (int sub = 0; sub < NSTEP; ++sub) {
      float t0 = tb + HS * (float)sub;
      for (int st = 0; st < 6; ++st) {
        // ---- stage state: as = a + h*sum(A[st][j]*ka_j); ys likewise (into LDS) ----
        if (st > 0) {
          for (int idx = tid; idx < NA; idx += NTHR) {
            float v = aS[idx];
            for (int j = 0; j < st; ++j) v += HS * STG_A[st][j] * kaB[j * NA + idx];
            asB[idx] = v;
          }
        }
        for (int idx = tid; idx < NY; idx += NTHR) {
          float v = yS[idx];
          for (int j = 0; j < st; ++j) v += HS * STG_A[st][j] * kyB[j * NY + idx];
          ysL[idx] = v;
        }
        __threadfence_block();
        __syncthreads();

        // ---- softmax rows of a_stage + yp = softmax(a)@y  ->  bf16 inp tile in LDS ----
        const float* __restrict__ aRead = (st == 0) ? aS : asB;
        float tstage = t0 + STG_C[st] * HS;
        for (int R = tid; R < NY; R += NTHR) {
          int b = R >> 6, i = R & 63;
          const float4* row = (const float4*)(aRead + b * HDIM * HDIM + i * HDIM);
          float mx = -1e30f;
          #pragma unroll
          for (int j = 0; j < 16; ++j) {
            float4 v = row[j];
            mx = fmaxf(mx, fmaxf(fmaxf(v.x, v.y), fmaxf(v.z, v.w)));
          }
          float den = 0.f, num = 0.f;
          #pragma unroll
          for (int j = 0; j < 16; ++j) {
            float4 v = row[j];
            const float* yb = ysL + b * HDIM + j * 4;
            float e0 = fexp2((v.x - mx) * LOG2E);
            float e1 = fexp2((v.y - mx) * LOG2E);
            float e2 = fexp2((v.z - mx) * LOG2E);
            float e3 = fexp2((v.w - mx) * LOG2E);
            den += e0 + e1 + e2 + e3;
            num += e0 * yb[0] + e1 * yb[1] + e2 * yb[2] + e3 * yb[3];
          }
          inpL[b * 96 + i] = f2bf(num * frcp(den));
        }
        if (tid < MT * 32) {  // t column + zero pad (K padded 65 -> 96)
          int b = tid >> 5, c = 64 + (tid & 31);
          inpL[b * 96 + c] = (c == 64) ? f2bf(tstage) : (unsigned short)0;
        }
        __syncthreads();

        // ---- MLPs via WMMA ----
        dense_pair_layer(wv, inpL, inpL, 96, 3, p.wF0, p.wG0, p.fb0, p.gb0, hfL[0], hgL[0]);
        __syncthreads();
        dense_pair_layer(wv, hfL[0], hgL[0], HDIM, 2, p.wF1, p.wG1, p.fb1, p.gb1, hfL[1], hgL[1]);
        __syncthreads();
        dense_pair_layer(wv, hfL[1], hgL[1], HDIM, 2, p.wF2, p.wG2, p.fb2, p.gb2, hfL[0], hgL[0]);
        __syncthreads();
        out_layers(wv, hfL[0], hgL[0], p.wF3, p.wG3, p.fb3, p.gb3, kyB + st * NY, kaB + st * NA);
        __threadfence_block();
        __syncthreads();
      }
      // ---- combine: state += h * sum(b_i * k_i) ----
      for (int idx = tid; idx < NA; idx += NTHR) {
        float acc = 0.f;
        #pragma unroll
        for (int j = 0; j < 6; ++j) acc += STG_B[j] * kaB[j * NA + idx];
        aS[idx] += HS * acc;
      }
      for (int idx = tid; idx < NY; idx += NTHR) {
        float acc = 0.f;
        #pragma unroll
        for (int j = 0; j < 6; ++j) acc += STG_B[j] * kyB[j * NY + idx];
        yS[idx] += HS * acc;
      }
      __threadfence_block();
      __syncthreads();
    }
    // ---- GRU cell + output write ----
    for (int idx = tid; idx < NY; idx += NTHR) ysL[idx] = yS[idx];
    __syncthreads();
    float res[2];
    #pragma unroll
    for (int q = 0; q < 2; ++q) {
      int R = tid + q * NTHR;
      int b = R >> 6, u = R & 63;
      const float* x = p.x_seq + ((size_t)(g * MT + b) * SEQ + s) * INPD;
      float ir = p.gru_b[u], iz = p.gru_b[64 + u], in = p.gru_b[128 + u];
      for (int j = 0; j < INPD; ++j) {
        float xv = x[j];
        ir += xv * p.gru_wi[u * INPD + j];
        iz += xv * p.gru_wi[(64 + u) * INPD + j];
        in += xv * p.gru_wi[(128 + u) * INPD + j];
      }
      float hr = 0.f, hz = 0.f, hn = 0.f;
      const float* yb = ysL + b * HDIM;
      for (int j = 0; j < HDIM; ++j) {
        float yv = yb[j];
        hr += yv * p.gru_wh[u * HDIM + j];
        hz += yv * p.gru_wh[(64 + u) * HDIM + j];
        hn += yv * p.gru_wh[(128 + u) * HDIM + j];
      }
      float r = sigmoid_f(ir + hr);
      float z = sigmoid_f(iz + hz);
      float n = tanh_f(in + r * (hn + p.gru_bn[u]));
      res[q] = n + z * (yb[u] - n);
    }
    __syncthreads();
    #pragma unroll
    for (int q = 0; q < 2; ++q) {
      int R = tid + q * NTHR;
      int b = R >> 6, u = R & 63;
      yS[R] = res[q];
      p.out[((size_t)(g * MT + b) * SEQ + s) * HDIM + u] = res[q];
    }
    __threadfence_block();
    __syncthreads();
  }
}

// ---------------- host ----------------
extern "C" void kernel_launch(void* const* d_in, const int* in_sizes, int n_in,
                              void* d_out, int out_size, void* d_ws, size_t ws_size,
                              hipStream_t stream) {
  (void)in_sizes; (void)n_in; (void)out_size;
  char* ws = (char*)d_ws;
  size_t off = 0;
  auto take = [&](size_t bytes) -> char* {
    char* p = ws + off;
    off = (off + bytes + 255) & ~(size_t)255;
    return p;
  };
  unsigned short* wF0 = (unsigned short*)take(3 * 4 * 512 * 2);
  unsigned short* wF1 = (unsigned short*)take(2 * 4 * 512 * 2);
  unsigned short* wF2 = (unsigned short*)take(2 * 4 * 512 * 2);
  unsigned short* wF3 = (unsigned short*)take(2 * 4 * 512 * 2);
  unsigned short* wG0 = (unsigned short*)take(3 * 4 * 512 * 2);
  unsigned short* wG1 = (unsigned short*)take(2 * 4 * 512 * 2);
  unsigned short* wG2 = (unsigned short*)take(2 * 4 * 512 * 2);
  unsigned short* wG3 = (unsigned short*)take(2 * 256 * 512 * 2);
  float* yS  = (float*)take((size_t)NGRP * MT * HDIM * 4);
  float* aS  = (float*)take((size_t)NGRP * MT * HDIM * HDIM * 4);
  float* asB = (float*)take((size_t)NGRP * MT * HDIM * HDIM * 4);
  float* kyB = (float*)take((size_t)NGRP * 6 * MT * HDIM * 4);
  float* kaB = (float*)take((size_t)NGRP * 6 * MT * HDIM * HDIM * 4);
  (void)ws_size;

  // inputs (setup_inputs order)
  const float* x_seq  = (const float*)d_in[0];
  const float* y0     = (const float*)d_in[1];
  const float* attn   = (const float*)d_in[2];
  const float* fW0    = (const float*)d_in[3];  const float* fb0 = (const float*)d_in[4];
  const float* fW1    = (const float*)d_in[5];  const float* fb1 = (const float*)d_in[6];
  const float* fW2    = (const float*)d_in[7];  const float* fb2 = (const float*)d_in[8];
  const float* fW3    = (const float*)d_in[9];  const float* fb3 = (const float*)d_in[10];
  const float* gW0    = (const float*)d_in[11]; const float* gb0 = (const float*)d_in[12];
  const float* gW1    = (const float*)d_in[13]; const float* gb1 = (const float*)d_in[14];
  const float* gW2    = (const float*)d_in[15]; const float* gb2 = (const float*)d_in[16];
  const float* gW3    = (const float*)d_in[17]; const float* gb3 = (const float*)d_in[18];
  const float* gru_wi = (const float*)d_in[19];
  const float* gru_wh = (const float*)d_in[20];
  const float* gru_b  = (const float*)d_in[21];
  const float* gru_bn = (const float*)d_in[22];

  // pack weights to bf16 WMMA-B layout (K padded to multiple of 32)
  pack_weight<<<24, 256, 0, stream>>>(fW0, wF0, 65, 3, 4);
  pack_weight<<<16, 256, 0, stream>>>(fW1, wF1, 64, 2, 4);
  pack_weight<<<16, 256, 0, stream>>>(fW2, wF2, 64, 2, 4);
  pack_weight<<<16, 256, 0, stream>>>(fW3, wF3, 64, 2, 4);
  pack_weight<<<24, 256, 0, stream>>>(gW0, wG0, 65, 3, 4);
  pack_weight<<<16, 256, 0, stream>>>(gW1, wG1, 64, 2, 4);
  pack_weight<<<16, 256, 0, stream>>>(gW2, wG2, 64, 2, 4);
  pack_weight<<<1024, 256, 0, stream>>>(gW3, wG3, 64, 2, 256);

  // init ODE state from y0 / attn (group-major layout == batch-major: straight copies)
  hipMemcpyAsync(yS, y0, (size_t)BATCH * HDIM * 4, hipMemcpyDeviceToDevice, stream);
  hipMemcpyAsync(aS, attn, (size_t)BATCH * HDIM * HDIM * 4, hipMemcpyDeviceToDevice, stream);

  AceParams p;
  p.x_seq = x_seq;
  p.fb0 = fb0; p.fb1 = fb1; p.fb2 = fb2; p.fb3 = fb3;
  p.gb0 = gb0; p.gb1 = gb1; p.gb2 = gb2; p.gb3 = gb3;
  p.gru_wi = gru_wi; p.gru_wh = gru_wh; p.gru_b = gru_b; p.gru_bn = gru_bn;
  p.wF0 = wF0; p.wF1 = wF1; p.wF2 = wF2; p.wF3 = wF3;
  p.wG0 = wG0; p.wG1 = wG1; p.wG2 = wG2; p.wG3 = wG3;
  p.yS = yS; p.aS = aS; p.asB = asB; p.kyB = kyB; p.kaB = kaB;
  p.out = (float*)d_out;

  ace_node_main<<<NGRP, NTHR, 0, stream>>>(p);
}
